// INR_45792941310797
// MI455X (gfx1250) — matrix operand order repaired and checked by reading, not compiled, not run on hardware
//
#include <hip/hip_runtime.h>
#include <hip/hip_bf16.h>

typedef __attribute__((ext_vector_type(16))) _Float16 v16h;
typedef __attribute__((ext_vector_type(8)))  _Float16 v8h;
typedef __attribute__((ext_vector_type(2)))  _Float16 v2h;
typedef __attribute__((ext_vector_type(8)))  float    v8f;
typedef __attribute__((ext_vector_type(4)))  unsigned int v4u;
typedef __attribute__((ext_vector_type(8)))  unsigned int v8u;
typedef _Float16 half_t;

#define DD     256      // feature dim
#define WSTR   264      // padded LDS row stride (halfs) for weights
#define ASTR   264      // padded LDS row stride (halfs) for activations
#define NWAVE  8        // waves per workgroup (256 threads, wave32)

// LDS sizes (halfs)
#define WTS_HALFS   (DD * WSTR)            // 67584 halfs = 135168 B
#define ACT_HALFS   (NWAVE * 16 * ASTR)    // 33792 halfs =  67584 B
#define SMEM_BYTES  ((WTS_HALFS + 2 * ACT_HALFS) * 2 + 128)   // + pool cells

// ---------------------------------------------------------------------------
// Prep: W1/W2 -> f16 transposed [n][k]; per-frame constants; dx/dy outputs
// ---------------------------------------------------------------------------
__global__ void inr_prep(const float* __restrict__ w1, const float* __restrict__ w2,
                         const int* __restrict__ sidx, const float* __restrict__ shift,
                         const float* __restrict__ rot, const float* __restrict__ colw,
                         const float* __restrict__ colb,
                         half_t* __restrict__ w1t, half_t* __restrict__ w2t,
                         float* __restrict__ consts, float* __restrict__ out_dxdy)
{
    int i = blockIdx.x * blockDim.x + threadIdx.x;
    if (i < DD * DD) {
        int n = i >> 8, k = i & 255;
        w1t[i] = (half_t)w1[k * DD + n];
    } else if (i < 2 * DD * DD) {
        int j = i - DD * DD;
        int n = j >> 8, k = j & 255;
        w2t[j] = (half_t)w2[k * DD + n];
    }
    if (i < 8) {
        int   idx = sidx[i];
        float a   = rot[idx];
        float dxv = shift[idx * 2 + 0], dyv = shift[idx * 2 + 1];
        float* cb = consts + i * 16;
        cb[0] = __cosf(a);
        cb[1] = __sinf(a);
        cb[2] = dxv;
        cb[3] = dyv;
        bool ident = (idx == 0);
        for (int c = 0; c < 3; ++c) {
            cb[4 + c] = ident ? 1.0f : colw[idx * 3 + c];
            cb[7 + c] = ident ? 0.0f : colb[idx * 3 + c];
        }
        out_dxdy[i]     = dxv;   // dx block of output tuple
        out_dxdy[8 + i] = dyv;   // dy block of output tuple
    }
}

// ---------------------------------------------------------------------------
// TDM: DMA the whole 256x256 f16 weight matrix into LDS with hardware row
// padding (512B rows padded by 16B -> WSTR=264 halfs). One instruction.
//   D# group0: count=1 | lds_addr | global_addr | type=2
//   D# group1: data_size=2B, pad_enable, pad_interval=128dw, pad_amount=4dw,
//              tensor/tile dims 256x256, dim0 stride 256
// ---------------------------------------------------------------------------
__device__ __forceinline__ void tdm_load_weights(const half_t* gsrc, unsigned lds_addr)
{
    unsigned long long ga = (unsigned long long)(uintptr_t)gsrc;
    v4u g0;
    g0[0] = 1u;                                                // count=1 (user D#)
    g0[1] = lds_addr;                                          // LDS byte address
    g0[2] = (unsigned)ga;                                      // global addr [31:0]
    g0[3] = (unsigned)((ga >> 32) & 0x01FFFFFFu) | 0x80000000u; // addr[56:32] | type=2
    v8u g1;
    g1[0] = (1u << 16)        // data_size = 1 -> 2-byte elements
          | (1u << 20)        // pad_enable
          | (6u << 22)        // pad_interval: 128 DWORDs (= one 512B row)
          | (3u << 25);       // pad_amount: 4 DWORDs (= 16B -> stride 264 halfs)
    g1[1] = (256u << 16);     // tensor_dim0 = 256 (low16)
    g1[2] = (256u << 16);     // tensor_dim0 hi=0 | tensor_dim1 = 256 (low16)
    g1[3] = (256u << 16);     // tensor_dim1 hi=0 | tile_dim0 = 256
    g1[4] = 256u;             // tile_dim1 = 256, tile_dim2 = 0
    g1[5] = 256u;             // tensor_dim0_stride low32 = 256 elements
    g1[6] = 0u;
    g1[7] = 0u;
    v4u g2; g2[0] = 1u; g2[1] = 0u; g2[2] = 0u; g2[3] = 0u;    // tensor_dim2=1
    v4u g3; g3[0] = 0u; g3[1] = 0u; g3[2] = 0u; g3[3] = 0u;
    asm volatile("tensor_load_to_lds %0, %1, %2, %3"
                 :: "s"(g0), "s"(g1), "s"(g2), "s"(g3)
                 : "memory");
}

// Async per-lane 16B global->LDS copy (ASYNCcnt path), GVS addressing.
__device__ __forceinline__ void async_ld_b128(unsigned lds_addr, const void* base,
                                              unsigned byte_off)
{
    asm volatile("global_load_async_to_lds_b128 %0, %1, %2"
                 :: "v"(lds_addr), "v"(byte_off), "s"(base)
                 : "memory");
}

// ---------------------------------------------------------------------------
// One 256->256 dense layer for a wave's 16-pixel tile using WMMA f16->f32.
// ---------------------------------------------------------------------------
__device__ __forceinline__ void mlp_layer(const half_t* __restrict__ ain,
                                          half_t* __restrict__ aout,
                                          const half_t* __restrict__ w,
                                          const float* __restrict__ bias,
                                          int lane)
{
    const int mrow  = lane & 15;       // A row / B-D column (within tile)
    const int hsel  = lane >> 4;       // half-wave select
    const int abase = mrow * ASTR + hsel * 8;   // A frag: + s*32 (+16 for hi 8)
    const int m_out = hsel << 3;                // D rows: m = r + 8*hsel

    for (int t = 0; t < 16; ++t) {              // 16 output N-tiles of 16
        v8f acc = {};
        const int     ncol = t * 16 + mrow;     // this lane's B column
        const half_t* wrow = w + ncol * WSTR + hsel * 8;
#pragma unroll
        for (int s = 0; s < 8; ++s) {           // K = 256 in steps of 32
            v8h alo = *(const v8h*)(ain + abase + s * 32);       // K: s*32+b..+7
            v8h ahi = *(const v8h*)(ain + abase + s * 32 + 16);  // K: s*32+16+b..
            v16h av = __builtin_shufflevector(alo, ahi,
                        0,1,2,3,4,5,6,7,8,9,10,11,12,13,14,15);
            v8h blo = *(const v8h*)(wrow + s * 32);
            v8h bhi = *(const v8h*)(wrow + s * 32 + 16);
            v16h bv = __builtin_shufflevector(blo, bhi,
                        0,1,2,3,4,5,6,7,8,9,10,11,12,13,14,15);
            acc = __builtin_amdgcn_wmma_f32_16x16x32_f16(
                      false, av, false, bv, (short)0, acc, false, false);
        }
        float bn = bias[ncol];
#pragma unroll
        for (int r = 0; r < 8; ++r) {
            float v = acc[r] + bn;
            v = fmaxf(v, 0.0f);
            aout[(m_out + r) * ASTR + t * 16 + mrow] = (half_t)v;
        }
    }
}

// ---------------------------------------------------------------------------
// Main fused kernel: WG = 8x16 pixel tile; wave = one 16-pixel row.
// ---------------------------------------------------------------------------
__global__ __launch_bounds__(256) void inr_main(
    const float* __restrict__ x,  const float* __restrict__ pw,
    const float* __restrict__ pb, const float* __restrict__ b1,
    const float* __restrict__ b2, const float* __restrict__ w3,
    const float* __restrict__ b3, const half_t* __restrict__ w1t,
    const half_t* __restrict__ w2t, const float* __restrict__ consts,
    float* __restrict__ out)
{
    extern __shared__ char smem[];
    half_t* WTS  = (half_t*)smem;
    half_t* A0   = WTS + WTS_HALFS;
    half_t* A1   = A0 + ACT_HALFS;
    float*  POOL = (float*)(A1 + ACT_HALFS);   // 24 f32 pool cells

    const int tid  = threadIdx.x;
    const int lane = tid & 31;
    const int wv   = tid >> 5;          // wave id = local pixel row (0..7)
    const int g    = blockIdx.x;
    const int b    = g >> 9;            // image (8 imgs * 512 WGs)
    const int rem  = g & 511;
    const int rt   = rem >> 4;          // row-tile (32 per image, 8 rows each)
    const int ct   = rem & 15;          // col-tile (16 per image, 16 cols each)

    const unsigned wts_base = (unsigned)(uintptr_t)WTS;   // LDS byte address

    if (tid < 24) POOL[tid] = 0.0f;

    // W1^T staging: single TDM DMA with hardware LDS row padding, issued by
    // wave 0 while all waves overlap the sin-projection compute below.
    if (wv == 0) {
        tdm_load_weights(w1t, wts_base);
    }

    // per-frame constants (uniform across WG)
    const float* cb = consts + b * 16;
    const float ca = cb[0], sa = cb[1], dxv = cb[2], dyv = cb[3];

    // h0 = sin(coords @ proj_w + proj_b) for this wave's 16 pixels
    {
        const int m  = lane & 15;
        const int hh = rt * 8 + wv;
        const int ww = ct * 16 + m;
        const int p  = (b * 256 + hh) * 256 + ww;
        const float cx = x[2 * p], cy = x[2 * p + 1];
        const float nx = ca * cx - sa * cy + dxv;
        const float ny = sa * cx + ca * cy + dyv;
        half_t* arow = A0 + (wv * 16 + m) * ASTR;
        const int dbase = (lane >> 4) * 128;    // lanes 0-15: d 0..127; 16-31: 128..255
        for (int j = 0; j < 128; j += 2) {
            const int d0 = dbase + j;
            float f0 = __sinf(nx * pw[d0]     + ny * pw[DD + d0]     + pb[d0]);
            float f1 = __sinf(nx * pw[d0 + 1] + ny * pw[DD + d0 + 1] + pb[d0 + 1]);
            v2h t2; t2[0] = (half_t)f0; t2[1] = (half_t)f1;
            *(v2h*)(arow + d0) = t2;
        }
    }
    if (wv == 0) {
        asm volatile("s_wait_tensorcnt 0x0" ::: "memory");   // W1 DMA complete
    }
    __syncthreads();

    // layer 1: A0 -> A1 with W1
    mlp_layer(A0 + wv * 16 * ASTR, A1 + wv * 16 * ASTR, WTS, b1, lane);
    __syncthreads();   // all waves done reading W1

    // W2^T staging: async global->LDS b128 copies (ASYNCcnt path), no VGPR data
    for (int i = tid; i < DD * 32; i += 256) {
        int n = i >> 5, sg = i & 31;
        async_ld_b128(wts_base + (unsigned)(n * WSTR + sg * 8) * 2,
                      (const void*)w2t, (unsigned)(n * DD + sg * 8) * 2);
    }
    asm volatile("s_wait_asynccnt 0x0" ::: "memory");
    __syncthreads();

    // layer 2: A1 -> A0 with W2
    mlp_layer(A1 + wv * 16 * ASTR, A0 + wv * 16 * ASTR, WTS, b2, lane);

    // layer 3 (256 -> 3) + pool accumulation (own tile only; LDS float atomics)
    {
        const int m  = lane & 15;
        const int kb = (lane >> 4) * 128;       // split K across half-waves
        const half_t* arow = A0 + (wv * 16 + m) * ASTR + kb;
        float p0 = 0.f, p1 = 0.f, p2 = 0.f;
        for (int j = 0; j < 128; j += 8) {
            v8h av = *(const v8h*)(arow + j);
#pragma unroll
            for (int e = 0; e < 8; ++e) {
                const float f = (float)av[e];
                const int   k = kb + j + e;
                p0 += f * w3[k * 3 + 0];
                p1 += f * w3[k * 3 + 1];
                p2 += f * w3[k * 3 + 2];
            }
        }
        const int blk = ((wv >> 2) << 2) + (m >> 2);   // 4x4 pool block id (0..7)
        atomicAdd(&POOL[blk * 3 + 0], p0);
        atomicAdd(&POOL[blk * 3 + 1], p1);
        atomicAdd(&POOL[blk * 3 + 2], p2);
    }
    __syncthreads();

    // finalize: mean, +b3, color scale/bias, store pooled pixel
    if (tid < 24) {
        const int blk = tid / 3, c = tid - blk * 3;
        float v = POOL[tid] * (1.0f / 16.0f) + b3[c];
        v = v * cb[4 + c] + cb[7 + c];
        const int ho = rt * 2 + (blk >> 2);
        const int wo = ct * 4 + (blk & 3);
        out[((b * 64 + ho) * 64 + wo) * 3 + c] = v;
    }
}

// ---------------------------------------------------------------------------
extern "C" void kernel_launch(void* const* d_in, const int* in_sizes, int n_in,
                              void* d_out, int out_size, void* d_ws, size_t ws_size,
                              hipStream_t stream) {
    (void)in_sizes; (void)n_in; (void)out_size; (void)ws_size;
    const float* x     = (const float*)d_in[0];
    const int*   sidx  = (const int*)d_in[1];
    const float* shift = (const float*)d_in[2];
    const float* rot   = (const float*)d_in[3];
    const float* pw    = (const float*)d_in[4];
    const float* pb    = (const float*)d_in[5];
    const float* w1    = (const float*)d_in[6];
    const float* b1    = (const float*)d_in[7];
    const float* w2    = (const float*)d_in[8];
    const float* b2    = (const float*)d_in[9];
    const float* w3    = (const float*)d_in[10];
    const float* b3    = (const float*)d_in[11];
    const float* colw  = (const float*)d_in[12];
    const float* colb  = (const float*)d_in[13];
    float* out = (float*)d_out;

    half_t* w1t    = (half_t*)d_ws;
    half_t* w2t    = w1t + DD * DD;
    float*  consts = (float*)((char*)d_ws + 2 * DD * DD * sizeof(half_t));

    inr_prep<<<dim3(512), dim3(256), 0, stream>>>(
        w1, w2, sidx, shift, rot, colw, colb, w1t, w2t, consts, out + 8 * 64 * 64 * 3);

    (void)hipFuncSetAttribute((const void*)inr_main,
                              hipFuncAttributeMaxDynamicSharedMemorySize,
                              (int)SMEM_BYTES);
    inr_main<<<dim3(4096), dim3(256), SMEM_BYTES, stream>>>(
        x, pw, pb, b1, b2, w3, b3, w1t, w2t, consts, out);
}
